// SentimentAwareAttention_70265664963030
// MI455X (gfx1250) — matrix operand order repaired and checked by reading, not compiled, not run on hardware
//
#include <hip/hip_runtime.h>
#include <hip/hip_bf16.h>

typedef __attribute__((ext_vector_type(16))) _Float16 v16h;
typedef __attribute__((ext_vector_type(8)))  float    v8f;

#define D_MODEL  768
#define N_HEADS  12
#define HEAD_DIM 64
#define B_SZ     2
#define SEQ      2048
#define M_TOTAL  (B_SZ * SEQ)   // 4096

#define BT_STRIDE 40   // transposed W tile row stride (halfs); 80B, 16B aligned
#define VT_STRIDE 72   // transposed V tile row stride (halfs); 144B, 16B aligned

// ---------------------------------------------------------------------------
// Kernel 1: Y = x @ W + bias, fp32 in, f16 out in [B*H][S][HEAD_DIM] layout.
// W tile is stored TRANSPOSED in LDS so B-operand reads are contiguous 16-half
// chunks (2 x ds_load_b128 instead of 16 scalar loads + v_mov_b16 packing).
// ---------------------------------------------------------------------------
__global__ __launch_bounds__(128) void qkv_proj_kernel(
    const float* __restrict__ x, const float* __restrict__ W,
    const float* __restrict__ bias, _Float16* __restrict__ out)
{
    __shared__ __align__(16) _Float16 ldsA[64 * 32];          // x tile (M x K)
    __shared__ __align__(16) _Float16 ldsBt[64 * BT_STRIDE];  // W tile (N x K)

    const int tid  = threadIdx.x;
    const int wave = tid >> 5, lane = tid & 31;
    const int l15  = lane & 15, hb = lane >> 4;
    const int m0   = blockIdx.x * 64;
    const int n0   = blockIdx.y * 64;          // == head * HEAD_DIM
    const int kb2  = hb * 8;                   // unused; A and B operands both use kbB
    const int kbB  = hb * 16;                  // B-operand K base per half-wave

    v8f acc[4] = {};

    for (int kk = 0; kk < D_MODEL; kk += 32) {
        // stage A: x[m0..+63][kk..+31], float4 vectorized, convert to f16
        #pragma unroll
        for (int it = 0; it < 4; ++it) {
            const int idx = tid + it * 128;
            const int row = idx >> 3, c4 = (idx & 7) * 4;
            const float4 v = *(const float4*)(x + (size_t)(m0 + row) * D_MODEL + kk + c4);
            _Float16* dst = ldsA + row * 32 + c4;
            dst[0] = (_Float16)v.x; dst[1] = (_Float16)v.y;
            dst[2] = (_Float16)v.z; dst[3] = (_Float16)v.w;
        }
        // stage B transposed: ldsBt[n][k] = W[kk+k][n0+n]
        #pragma unroll
        for (int it = 0; it < 4; ++it) {
            const int idx = tid + it * 128;
            const int k = idx >> 4, c4 = (idx & 15) * 4;
            const float4 v = *(const float4*)(W + (size_t)(kk + k) * D_MODEL + n0 + c4);
            _Float16* dst = ldsBt + c4 * BT_STRIDE + k;
            dst[0]             = (_Float16)v.x;
            dst[BT_STRIDE]     = (_Float16)v.y;
            dst[2 * BT_STRIDE] = (_Float16)v.z;
            dst[3 * BT_STRIDE] = (_Float16)v.w;
        }
        __syncthreads();

        // A operand (16x32) for this wave, contiguous k = 16*hb + i, same order as B
        v16h a;
        const int arow = wave * 16 + l15;
        #pragma unroll
        for (int i = 0; i < 16; ++i) {
            int k = kbB + i;
            a[i] = ldsA[arow * 32 + k];
        }
        #pragma unroll
        for (int t = 0; t < 4; ++t) {
            v16h b;   // contiguous 16 halfs from transposed tile
            #pragma unroll
            for (int i = 0; i < 16; ++i)
                b[i] = ldsBt[(t * 16 + l15) * BT_STRIDE + kbB + i];
            acc[t] = __builtin_amdgcn_wmma_f32_16x16x32_f16(
                false, a, false, b, (short)0, acc[t], false, false);
        }
        __syncthreads();
    }

    // epilogue: + bias, convert to f16, scatter to [b*12+h][s][d]
    const int h = blockIdx.y;
    #pragma unroll
    for (int t = 0; t < 4; ++t) {
        const int d   = t * 16 + l15;
        const float bb = bias[n0 + d];
        #pragma unroll
        for (int r = 0; r < 8; ++r) {
            const int gm = m0 + wave * 16 + r + hb * 8;   // global row (b*S + s)
            const int b  = gm >> 11, s = gm & (SEQ - 1);
            out[((size_t)(b * N_HEADS + h) * SEQ + s) * HEAD_DIM + d] =
                (_Float16)(acc[t][r] + bb);
        }
    }
}

// ---------------------------------------------------------------------------
// Kernel 2: flash attention. One WG (128 thr, 4 waves) per (b,h) x 64-query
// block. K chunk is brought in with GLOBAL_LOAD_ASYNC_TO_LDS_B128 (ASYNCcnt
// path, no VGPR round-trip); V chunk is staged transposed so the AV B-operand
// reads are contiguous. Sentiment bias is constant along the softmax axis ->
// exactly zero effect on the output, so it is skipped.
// ---------------------------------------------------------------------------
__global__ __launch_bounds__(128) void flash_attn_kernel(
    const _Float16* __restrict__ Qf, const _Float16* __restrict__ Kf,
    const _Float16* __restrict__ Vf, float* __restrict__ attn_out)
{
    __shared__ __align__(16) _Float16 ldsK[64 * 64];          // K row-major
    __shared__ __align__(16) _Float16 ldsVt[64 * VT_STRIDE];  // V transposed [d][j]
    __shared__ __align__(16) _Float16 ldsP[4 * 16 * 64];      // per-wave P staging

    const int tid  = threadIdx.x;
    const int wave = tid >> 5, lane = tid & 31;
    const int l15  = lane & 15, hb = lane >> 4;
    const int bh   = blockIdx.y;
    const int q0   = blockIdx.x * 64;
    const int kb2  = hb * 8, kbB = hb * 16;

    const _Float16* Qg = Qf + (size_t)bh * SEQ * HEAD_DIM;
    const _Float16* Kg = Kf + (size_t)bh * SEQ * HEAD_DIM;
    const _Float16* Vg = Vf + (size_t)bh * SEQ * HEAD_DIM;

    // Q operand tiles in registers (A layout), d-chunks [0,32) and [32,64)
    v16h qa[2];
    {
        const int row = q0 + wave * 16 + l15;
        #pragma unroll
        for (int c = 0; c < 2; ++c)
            #pragma unroll
            for (int i = 0; i < 16; ++i) {
                int dd = c * 32 + kbB + i;
                qa[c][i] = Qg[(size_t)row * HEAD_DIM + dd];
            }
    }

    float m_r[8], l_r[8];
    #pragma unroll
    for (int r = 0; r < 8; ++r) { m_r[r] = -1e30f; l_r[r] = 0.f; }
    v8f o[4] = {};

    for (int kc = 0; kc < SEQ; kc += 64) {
        // ---- K chunk: async DMA straight into LDS (64x64 halfs = 512 x b128)
        {
            const _Float16* gk = Kg + (size_t)kc * HEAD_DIM;
#if defined(__HIP_DEVICE_COMPILE__) && defined(__gfx1250__)
            #pragma unroll
            for (int it = 0; it < 4; ++it) {
                const int i = tid + it * 128;
                const unsigned lk = (unsigned)(size_t)(const void*)(ldsK + i * 8);
                const void* ak = (const void*)(gk + i * 8);
                asm volatile("global_load_async_to_lds_b128 %0, %1, off"
                             :: "v"(lk), "v"(ak) : "memory");
            }
            asm volatile("s_wait_asynccnt 0" ::: "memory");
#else
            for (int i = tid; i < 1024; i += 128)
                ((uint2*)ldsK)[i] = ((const uint2*)gk)[i];
#endif
        }
        // ---- V chunk: staged transposed ldsVt[d][j] for contiguous B reads
        {
            const uint2* sv = (const uint2*)(Vg + (size_t)kc * HEAD_DIM);
            #pragma unroll
            for (int it = 0; it < 8; ++it) {
                const int i = tid + it * 128;     // uint2 index: 4 halfs along d
                const int j = i >> 4;             // key row
                const int d = (i & 15) * 4;       // d offset
                union { uint2 u; _Float16 h[4]; } cv; cv.u = sv[i];
                ldsVt[(d + 0) * VT_STRIDE + j] = cv.h[0];
                ldsVt[(d + 1) * VT_STRIDE + j] = cv.h[1];
                ldsVt[(d + 2) * VT_STRIDE + j] = cv.h[2];
                ldsVt[(d + 3) * VT_STRIDE + j] = cv.h[3];
            }
        }
        if (kc + 64 < SEQ) {   // prefetch next chunk (global_prefetch_b8)
            __builtin_prefetch(Kg + (size_t)(kc + 64) * HEAD_DIM + tid * 32, 0, 3);
            __builtin_prefetch(Vg + (size_t)(kc + 64) * HEAD_DIM + tid * 32, 0, 3);
        }
        __syncthreads();

        // scores: 16x64 per wave = 4 tiles x 2 k-chunks of WMMA
        v8f st[4];
        #pragma unroll
        for (int jt = 0; jt < 4; ++jt) {
            v8f s = {};
            #pragma unroll
            for (int c = 0; c < 2; ++c) {
                v16h b;   // B[d][j] = K[j][d]: contiguous along d in ldsK
                #pragma unroll
                for (int i = 0; i < 16; ++i)
                    b[i] = ldsK[(jt * 16 + l15) * 64 + c * 32 + kbB + i];
                s = __builtin_amdgcn_wmma_f32_16x16x32_f16(
                    false, qa[c], false, b, (short)0, s, false, false);
            }
            st[jt] = s;
        }

        // online softmax (scale = 1/sqrt(64) = 0.125)
        float alpha[8], rowsum[8];
        #pragma unroll
        for (int r = 0; r < 8; ++r) {
            float mx = st[0][r];
            #pragma unroll
            for (int jt = 1; jt < 4; ++jt) mx = fmaxf(mx, st[jt][r]);
            mx *= 0.125f;
            mx = fmaxf(mx, __shfl_xor(mx, 1, 32));
            mx = fmaxf(mx, __shfl_xor(mx, 2, 32));
            mx = fmaxf(mx, __shfl_xor(mx, 4, 32));
            mx = fmaxf(mx, __shfl_xor(mx, 8, 32));
            const float mn = fmaxf(m_r[r], mx);
            alpha[r]  = __expf(m_r[r] - mn);
            m_r[r]    = mn;
            rowsum[r] = 0.f;
        }

        // P = exp(s*scale - m), staged to LDS as f16 in row-major per wave
        #pragma unroll
        for (int jt = 0; jt < 4; ++jt)
            #pragma unroll
            for (int r = 0; r < 8; ++r) {
                const float p = __expf(st[jt][r] * 0.125f - m_r[r]);
                rowsum[r] += p;
                ldsP[wave * (16 * 64) + (r + hb * 8) * 64 + jt * 16 + l15] = (_Float16)p;
            }

        #pragma unroll
        for (int r = 0; r < 8; ++r) {
            float s = rowsum[r];
            s += __shfl_xor(s, 1, 32); s += __shfl_xor(s, 2, 32);
            s += __shfl_xor(s, 4, 32); s += __shfl_xor(s, 8, 32);
            l_r[r] = l_r[r] * alpha[r] + s;
            #pragma unroll
            for (int t = 0; t < 4; ++t) o[t][r] *= alpha[r];
        }

        // O += P @ V
        #pragma unroll
        for (int c = 0; c < 2; ++c) {
            v16h pa;   // A-operand row: one contiguous 16-half run, k = 16*hb + i
            #pragma unroll
            for (int i = 0; i < 16; ++i) {
                int j = c * 32 + kbB + i;
                pa[i] = ldsP[wave * (16 * 64) + l15 * 64 + j];
            }
            #pragma unroll
            for (int t = 0; t < 4; ++t) {
                v16h b;   // contiguous 16 halfs from transposed V tile
                #pragma unroll
                for (int i = 0; i < 16; ++i)
                    b[i] = ldsVt[(t * 16 + l15) * VT_STRIDE + c * 32 + kbB + i];
                o[t] = __builtin_amdgcn_wmma_f32_16x16x32_f16(
                    false, pa, false, b, (short)0, o[t], false, false);
            }
        }
        __syncthreads();
    }

    // epilogue: normalize by row sum, write fp32 [b][s][h*64+d]
    const int b = bh / N_HEADS, h = bh % N_HEADS;
    #pragma unroll
    for (int r = 0; r < 8; ++r) {
        const float inv  = 1.f / l_r[r];
        const int s_idx  = q0 + wave * 16 + r + hb * 8;
        #pragma unroll
        for (int t = 0; t < 4; ++t)
            attn_out[(size_t)(b * SEQ + s_idx) * D_MODEL + h * HEAD_DIM + t * 16 + l15] =
                o[t][r] * inv;
    }
}

// ---------------------------------------------------------------------------
// Kernel 3: y = attn + x ; out = LayerNorm(y) * g + beta.  One block per row.
// ---------------------------------------------------------------------------
__global__ __launch_bounds__(256) void resid_ln_kernel(
    const float* __restrict__ attn, const float* __restrict__ x,
    const float* __restrict__ g, const float* __restrict__ beta,
    float* __restrict__ out)
{
    __shared__ float rs[256], rss[256];
    const int row = blockIdx.x;
    const int tid = threadIdx.x;
    const float* a  = attn + (size_t)row * D_MODEL;
    const float* xr = x    + (size_t)row * D_MODEL;

    float y[3], s = 0.f, ss = 0.f;
    #pragma unroll
    for (int i = 0; i < 3; ++i) {
        const int c = tid + i * 256;
        const float v = a[c] + xr[c];
        y[i] = v; s += v; ss += v * v;
    }
    rs[tid] = s; rss[tid] = ss;
    __syncthreads();
    for (int off = 128; off > 0; off >>= 1) {
        if (tid < off) { rs[tid] += rs[tid + off]; rss[tid] += rss[tid + off]; }
        __syncthreads();
    }
    const float mean = rs[0]  * (1.f / D_MODEL);
    const float var  = rss[0] * (1.f / D_MODEL) - mean * mean;
    const float rstd = rsqrtf(var + 1e-5f);
    #pragma unroll
    for (int i = 0; i < 3; ++i) {
        const int c = tid + i * 256;
        out[(size_t)row * D_MODEL + c] = (y[i] - mean) * rstd * g[c] + beta[c];
    }
}

// ---------------------------------------------------------------------------
extern "C" void kernel_launch(void* const* d_in, const int* in_sizes, int n_in,
                              void* d_out, int out_size, void* d_ws, size_t ws_size,
                              hipStream_t stream) {
    (void)in_sizes; (void)n_in; (void)out_size; (void)ws_size;

    const float* x    = (const float*)d_in[0];
    // d_in[1] sentiment_context, d_in[8] Ws, d_in[9] bs: per-(b,h) additive
    // logit bias is constant along the softmax axis -> mathematically a no-op.
    const float* Wq   = (const float*)d_in[2];
    const float* bq   = (const float*)d_in[3];
    const float* Wk   = (const float*)d_in[4];
    const float* bk   = (const float*)d_in[5];
    const float* Wv   = (const float*)d_in[6];
    const float* bv   = (const float*)d_in[7];
    const float* ln_g = (const float*)d_in[10];
    const float* ln_b = (const float*)d_in[11];

    // workspace layout
    _Float16* Qf = (_Float16*)d_ws;                       // 4096*768 halfs
    _Float16* Kf = Qf + (size_t)M_TOTAL * D_MODEL;
    _Float16* Vf = Kf + (size_t)M_TOTAL * D_MODEL;
    float* attn  = (float*)(Vf + (size_t)M_TOTAL * D_MODEL);  // 4096*768 f32

    const dim3 gq(M_TOTAL / 64, D_MODEL / 64);            // 64 x 12
    qkv_proj_kernel<<<gq, 128, 0, stream>>>(x, Wq, bq, Qf);
    qkv_proj_kernel<<<gq, 128, 0, stream>>>(x, Wk, bk, Kf);
    qkv_proj_kernel<<<gq, 128, 0, stream>>>(x, Wv, bv, Vf);

    flash_attn_kernel<<<dim3(SEQ / 64, B_SZ * N_HEADS), 128, 0, stream>>>(
        Qf, Kf, Vf, attn);

    resid_ln_kernel<<<M_TOTAL, 256, 0, stream>>>(attn, x, ln_g, ln_b, (float*)d_out);
}
